// Attention_55276229099869
// MI455X (gfx1250) — compile-verified, matrix operand order checked by reading, not run on hardware
//
#include <hip/hip_runtime.h>
#include <hip/hip_bf16.h>

typedef __attribute__((ext_vector_type(16))) _Float16 v16h;
typedef __attribute__((ext_vector_type(8)))  _Float16 v8h;
typedef __attribute__((ext_vector_type(8)))  float    v8f;

#define HID   1024
#define S_LEN 2048
#define NH    16
#define HD    64
#define NTOK  4096   // B * S

#if __has_builtin(__builtin_amdgcn_exp2f)
#define EXP2F(x) __builtin_amdgcn_exp2f(x)
#else
#define EXP2F(x) exp2f(x)
#endif

#define LOG2E 1.44269504088896f

// Build a v16h fragment from two 16-byte chunks (global b128 or ds b128 loads).
static __device__ __forceinline__ v16h ld_frag(const _Float16* p0, const _Float16* p1) {
  v8h a = *(const v8h*)p0;
  v8h b = *(const v8h*)p1;
  return __builtin_shufflevector(a, b, 0,1,2,3,4,5,6,7,8,9,10,11,12,13,14,15);
}

// ---------------- weight f32 -> f16 convert ----------------
__global__ void w2h_kernel(const float* __restrict__ w, _Float16* __restrict__ o) {
  int i = blockIdx.x * blockDim.x + threadIdx.x;
  o[i] = (_Float16)w[i];
}

// ---------------- LayerNorm -> f16 ----------------
__global__ void ln_kernel(const float* __restrict__ x, const float* __restrict__ gamma,
                          const float* __restrict__ beta, _Float16* __restrict__ out) {
  int row = blockIdx.x;
  int tid = threadIdx.x;
  const float4* xr = (const float4*)(x + (size_t)row * HID);
  float4 v = xr[tid];                              // 256 threads * 4 = 1024
  float s  = v.x + v.y + v.z + v.w;
  float s2 = v.x*v.x + v.y*v.y + v.z*v.z + v.w*v.w;
  for (int off = 16; off; off >>= 1) {
    s  += __shfl_down(s,  off, 32);
    s2 += __shfl_down(s2, off, 32);
  }
  __shared__ float sb[8], qb[8];
  if ((tid & 31) == 0) { sb[tid >> 5] = s; qb[tid >> 5] = s2; }
  __syncthreads();
  float ts = 0.f, tq = 0.f;
  for (int i = 0; i < 8; ++i) { ts += sb[i]; tq += qb[i]; }
  float mean = ts * (1.0f / HID);
  float var  = tq * (1.0f / HID) - mean * mean;
  float rstd = rsqrtf(var + 1e-12f);
  int c = tid * 4;
  _Float16* orow = out + (size_t)row * HID;
  float vv[4] = {v.x, v.y, v.z, v.w};
#pragma unroll
  for (int j = 0; j < 4; ++j)
    orow[c + j] = (_Float16)(((vv[j] - mean) * rstd) * gamma[c + j] + beta[c + j]);
}

// ---------------- generic WMMA GEMM: out = A @ W^T + bias ----------------
// A: [4096, 1024] f16 row-major.  Wh: [1024(n), 1024(k)] f16 (B[k,n]=Wh[n,k], k contiguous).
// One wave computes a 32x64 tile (2 A-frags share 4 B-frags).
// mode 0: q (scale * log2e, head-major)   mode 1: k (scale, head-major)
// mode 2: v transposed [b,h,d,s]          mode 3: f32 out (final projection)
__global__ void proj_kernel(const _Float16* __restrict__ A, const _Float16* __restrict__ Wh,
                            const float* __restrict__ bias, _Float16* __restrict__ o16,
                            float* __restrict__ o32, int mode) {
  const int lane = threadIdx.x;
  const int nlo  = lane & 15;    // A: row within tile / B: column within tile / C: N
  const int g    = lane >> 4;    // k-chunk select; C rows +8
  const int m0 = blockIdx.x * 32;
  const int n0 = blockIdx.y * 64;

  const _Float16* arow0 = A + (size_t)(m0 + nlo) * HID;
  const _Float16* arow1 = arow0 + (size_t)16 * HID;
  const _Float16* brow[4];
#pragma unroll
  for (int t = 0; t < 4; ++t)
    brow[t] = Wh + (size_t)(n0 + t*16 + nlo) * HID;

  v8f acc[2][4] = {};

  for (int kk = 0; kk < HID; kk += 32) {
    // Batch all loads for this k-step so one wait covers the 8-WMMA chain.
    v16h aF0 = ld_frag(arow0 + kk + g*8, arow0 + kk + 16 + g*8);
    v16h aF1 = ld_frag(arow1 + kk + g*8, arow1 + kk + 16 + g*8);
    v16h bF[4];
#pragma unroll
    for (int t = 0; t < 4; ++t)
      bF[t] = ld_frag(brow[t] + kk + g*8, brow[t] + kk + 16 + g*8);
    // Warm L0/L2 for the next k-chunk while the WMMA chain runs.
#pragma unroll
    for (int t = 0; t < 4; ++t)
      __builtin_prefetch(brow[t] + kk + 32, 0, 0);
    __builtin_prefetch(arow0 + kk + 32, 0, 0);
    __builtin_prefetch(arow1 + kk + 32, 0, 0);
#pragma unroll
    for (int t = 0; t < 4; ++t)
      acc[0][t] = __builtin_amdgcn_wmma_f32_16x16x32_f16(
          false, aF0, false, bF[t], (short)0, acc[0][t], false, false);
#pragma unroll
    for (int t = 0; t < 4; ++t)
      acc[1][t] = __builtin_amdgcn_wmma_f32_16x16x32_f16(
          false, aF1, false, bF[t], (short)0, acc[1][t], false, false);
  }

  const float qkscale = (mode == 0) ? 0.35355339059327f * LOG2E : 0.35355339059327f;
#pragma unroll
  for (int mi = 0; mi < 2; ++mi) {
#pragma unroll
    for (int t = 0; t < 4; ++t) {
#pragma unroll
      for (int r = 0; r < 8; ++r) {
        int m = mi*16 + r + 8*g;
        int token = m0 + m;
        int n = n0 + t*16 + nlo;
        float val = acc[mi][t][r] + bias[n];
        int b = token >> 11, s = token & (S_LEN - 1);
        if (mode <= 1) {               // q or k: head-major [b,h,s,d]
          int head = n >> 6, d = n & 63;
          o16[((size_t)((b*NH + head)*S_LEN + s))*HD + d] = (_Float16)(val * qkscale);
        } else if (mode == 2) {        // v: transposed [b,h,d,s]
          int head = n >> 6, d = n & 63;
          o16[((size_t)((b*NH + head)*HD + d))*S_LEN + s] = (_Float16)val;
        } else {                       // final projection: f32 output
          o32[(size_t)token * HID + n] = val;
        }
      }
    }
  }
}

// ---------------- flash attention: one wave per (b, h, 16-row q tile) ----------------
// Scores come out of WMMA already multiplied by log2(e) (folded into q at projection),
// so the online softmax runs entirely in base 2 (v_exp_f32).
__global__ void attn_kernel(const _Float16* __restrict__ q, const _Float16* __restrict__ k,
                            const _Float16* __restrict__ vt, const float* __restrict__ mask,
                            _Float16* __restrict__ ctx) {
  const int lane = threadIdx.x;
  const int nlo = lane & 15;
  const int g   = lane >> 4;
  const int q0 = blockIdx.x * 16;
  const int h  = blockIdx.y;
  const int b  = blockIdx.z;

  const _Float16* qh = q  + ((size_t)(b*NH + h) * S_LEN) * HD;
  const _Float16* kh = k  + ((size_t)(b*NH + h) * S_LEN) * HD;
  const _Float16* vh = vt + ((size_t)(b*NH + h) * HD) * S_LEN;
  const float* mb = mask + (size_t)b * S_LEN;

  const _Float16* qrow = qh + (size_t)(q0 + nlo) * HD;
  v16h aQ0 = ld_frag(qrow + g*8,      qrow + 16 + g*8);   // d = 0..31
  v16h aQ1 = ld_frag(qrow + 32 + g*8, qrow + 48 + g*8);   // d = 32..63

  float mrow[8], lrow[8];
#pragma unroll
  for (int r = 0; r < 8; ++r) { mrow[r] = -3.0e38f; lrow[r] = 0.f; }
  v8f accv[4] = {};

  __shared__ __align__(16) _Float16 pbuf[16 * 40];   // 16 rows x 32 cols, padded stride 40

  for (int kv = 0; kv < S_LEN; kv += 32) {
    const _Float16* k0 = kh + (size_t)(kv + nlo) * HD;
    const _Float16* k1 = k0 + 16 * HD;
    // Batch all 4 K-fragment loads, then chain 4 WMMAs behind one wait.
    v16h b00 = ld_frag(k0 + g*8,      k0 + 16 + g*8);
    v16h b01 = ld_frag(k0 + 32 + g*8, k0 + 48 + g*8);
    v16h b10 = ld_frag(k1 + g*8,      k1 + 16 + g*8);
    v16h b11 = ld_frag(k1 + 32 + g*8, k1 + 48 + g*8);
    __builtin_prefetch(k0 + 32 * HD, 0, 0);            // next kv block, this lane's rows
    __builtin_prefetch(k1 + 32 * HD, 0, 0);
    v8f s0 = {}, s1 = {};
    s0 = __builtin_amdgcn_wmma_f32_16x16x32_f16(false, aQ0, false, b00, (short)0, s0, false, false);
    s0 = __builtin_amdgcn_wmma_f32_16x16x32_f16(false, aQ1, false, b01, (short)0, s0, false, false);
    s1 = __builtin_amdgcn_wmma_f32_16x16x32_f16(false, aQ0, false, b10, (short)0, s1, false, false);
    s1 = __builtin_amdgcn_wmma_f32_16x16x32_f16(false, aQ1, false, b11, (short)0, s1, false, false);

    // additive mask, scaled into base-2 domain
    float am0 = -1000.0f * LOG2E * (1.0f - mb[kv + nlo]);
    float am1 = -1000.0f * LOG2E * (1.0f - mb[kv + 16 + nlo]);

    float p0[8], p1[8], alpha[8];
#pragma unroll
    for (int r = 0; r < 8; ++r) {
      float x0 = s0[r] + am0;
      float x1 = s1[r] + am1;
      float cmax = fmaxf(x0, x1);
      for (int off = 8; off; off >>= 1)            // row-wise max across 16 lanes
        cmax = fmaxf(cmax, __shfl_xor(cmax, off, 16));
      float mnew = fmaxf(mrow[r], cmax);
      alpha[r] = EXP2F(mrow[r] - mnew);
      p0[r] = EXP2F(x0 - mnew);
      p1[r] = EXP2F(x1 - mnew);
      float rs = p0[r] + p1[r];
      for (int off = 8; off; off >>= 1)            // row-wise sum
        rs += __shfl_xor(rs, off, 16);
      lrow[r] = lrow[r] * alpha[r] + rs;
      mrow[r] = mnew;
    }
#pragma unroll
    for (int t = 0; t < 4; ++t)
#pragma unroll
      for (int r = 0; r < 8; ++r)
        accv[t][r] *= alpha[r];

    // C-layout -> A-layout for P via LDS
#pragma unroll
    for (int r = 0; r < 8; ++r) {
      int row = r + 8*g;
      pbuf[row*40 + nlo]      = (_Float16)p0[r];
      pbuf[row*40 + 16 + nlo] = (_Float16)p1[r];
    }
    __syncthreads();
    v16h aP = ld_frag(&pbuf[nlo*40 + g*8], &pbuf[nlo*40 + 16 + g*8]);
    __syncthreads();

    // Batch all 4 V-fragment loads, then chain 4 WMMAs.
    v16h bV[4];
#pragma unroll
    for (int t = 0; t < 4; ++t) {
      const _Float16* vrow = vh + (size_t)(t*16 + nlo) * S_LEN + kv;  // Vt: contiguous in kv
      bV[t] = ld_frag(vrow + g*8, vrow + 16 + g*8);
      __builtin_prefetch(vrow + 32, 0, 0);
    }
#pragma unroll
    for (int t = 0; t < 4; ++t)
      accv[t] = __builtin_amdgcn_wmma_f32_16x16x32_f16(
          false, aP, false, bV[t], (short)0, accv[t], false, false);
  }

  float rinv[8];
#pragma unroll
  for (int r = 0; r < 8; ++r) rinv[r] = 1.0f / lrow[r];
#pragma unroll
  for (int t = 0; t < 4; ++t)
#pragma unroll
    for (int r = 0; r < 8; ++r) {
      int m = r + 8*g;
      int token = b * S_LEN + q0 + m;
      int col = h*HD + t*16 + nlo;
      ctx[(size_t)token * HID + col] = (_Float16)(accv[t][r] * rinv[r]);
    }
}

// ---------------- host side ----------------
extern "C" void kernel_launch(void* const* d_in, const int* in_sizes, int n_in,
                              void* d_out, int out_size, void* d_ws, size_t ws_size,
                              hipStream_t stream) {
  (void)in_sizes; (void)n_in; (void)out_size; (void)ws_size;
  const float* hs   = (const float*)d_in[0];
  const float* mask = (const float*)d_in[1];
  const float* Wq = (const float*)d_in[2];
  const float* bq = (const float*)d_in[3];
  const float* Wk = (const float*)d_in[4];
  const float* bk = (const float*)d_in[5];
  const float* Wv = (const float*)d_in[6];
  const float* bv = (const float*)d_in[7];
  const float* Wd = (const float*)d_in[8];
  const float* bd = (const float*)d_in[9];
  const float* gam = (const float*)d_in[10];
  const float* bet = (const float*)d_in[11];
  float* out = (float*)d_out;

  char* ws = (char*)d_ws;
  const size_t SZH = (size_t)NTOK * HID * sizeof(_Float16);  // 8 MB
  const size_t SZW = (size_t)HID * HID * sizeof(_Float16);   // 2 MB
  _Float16* h16  = (_Float16*)(ws);
  _Float16* q16  = (_Float16*)(ws + 1*SZH);
  _Float16* k16  = (_Float16*)(ws + 2*SZH);
  _Float16* vt16 = (_Float16*)(ws + 3*SZH);
  _Float16* c16  = (_Float16*)(ws + 4*SZH);
  _Float16* wq16 = (_Float16*)(ws + 5*SZH);
  _Float16* wk16 = (_Float16*)(ws + 5*SZH + 1*SZW);
  _Float16* wv16 = (_Float16*)(ws + 5*SZH + 2*SZW);
  _Float16* wd16 = (_Float16*)(ws + 5*SZH + 3*SZW);

  const int WN = HID * HID / 256;
  w2h_kernel<<<WN, 256, 0, stream>>>(Wq, wq16);
  w2h_kernel<<<WN, 256, 0, stream>>>(Wk, wk16);
  w2h_kernel<<<WN, 256, 0, stream>>>(Wv, wv16);
  w2h_kernel<<<WN, 256, 0, stream>>>(Wd, wd16);

  ln_kernel<<<NTOK, 256, 0, stream>>>(hs, gam, bet, h16);

  dim3 pg(NTOK / 32, HID / 64);
  proj_kernel<<<pg, 32, 0, stream>>>(h16, wq16, bq, q16,  nullptr, 0);
  proj_kernel<<<pg, 32, 0, stream>>>(h16, wk16, bk, k16,  nullptr, 1);
  proj_kernel<<<pg, 32, 0, stream>>>(h16, wv16, bv, vt16, nullptr, 2);

  attn_kernel<<<dim3(S_LEN / 16, NH, 2), 32, 0, stream>>>(q16, k16, vt16, mask, c16);

  proj_kernel<<<pg, 32, 0, stream>>>(c16, wd16, bd, nullptr, out, 3);
}
